// CausalGraphEmbedder_87058987090399
// MI455X (gfx1250) — matrix operand order repaired and checked by reading, not compiled, not run on hardware
//
#include <hip/hip_runtime.h>
#include <hip/hip_bf16.h>

// ---------------------------------------------------------------------------
// GAT x2 + LayerNorm + mean for MI455X (gfx1250, wave32, WMMA + TDM).
// - Fused flash-style attention: [H,N,N] never materialized; adj (64 MB, the
//   real HBM stream) double-buffer staged via the Tensor Data Mover into LDS.
// - Wh f16 stored in WMMA-B-operand packed order by the GEMM epilogue, so the
//   attention B operand is one contiguous 32B load (L2-resident, 192MB L2).
// - P@Wh and h@W on v_wmma_f32_16x16x32_f16 with f32 accumulation; softmax
//   statistics (shift, Z) kept in f32; P <= 1 so f16-safe.
// ---------------------------------------------------------------------------

#define GAT_N     4096
#define GAT_FIN   128
#define GAT_EMB   128
#define GAT_H     4
#define GAT_ALPHA 0.2f

typedef __attribute__((ext_vector_type(16))) _Float16 v16h;
typedef __attribute__((ext_vector_type(8)))  _Float16 v8h;
typedef __attribute__((ext_vector_type(8)))  float    v8f;
typedef __attribute__((ext_vector_type(4)))  unsigned int u32x4;
typedef __attribute__((ext_vector_type(8)))  int      i32x8;
typedef __attribute__((ext_vector_type(4)))  int      i32x4;

#if defined(__has_builtin)
#if __has_builtin(__builtin_amdgcn_tensor_load_to_lds) && \
    __has_builtin(__builtin_amdgcn_s_wait_tensorcnt)
#define GAT_HAVE_TDM 1
#endif
#endif
#ifndef GAT_HAVE_TDM
#warning "gfx1250 TDM builtins not found: falling back to cooperative LDS staging"
#endif

// ---- WMMA operand index maps (CDNA5 ISA 7.12.2, wave32) -------------------
// A operand: 16x32 f16. lane l: row m=l&15, kb=(l<16)?0:8; per-lane elements
// are two contiguous 8-runs: k = kb+e (e<8), kb+16+(e-8) (e>=8).
// B operand: 32x16 f16. lane l holds col n=l&15; k = kb+e (e<8), kb+16+(e-8) (e>=8).
__device__ __forceinline__ int b_kk(int lane, int e) {
  return ((lane < 16) ? 0 : 8) + e + ((e < 8) ? 0 : 8);
}
// C/D: 16x16 f32, vgpr r: row = ((lane<16)?0:8)+r, col = lane&15.

// ---- Tensor Data Mover: 2-D tile load Global -> LDS (D# per ISA ch.8) -----
#ifdef GAT_HAVE_TDM
__device__ __forceinline__ void tdm_load_2d(unsigned int lds_off,
                                            const void* gaddr,
                                            unsigned int elem_code,  // 0:1B 1:2B 2:4B 3:8B
                                            unsigned int tile_w, unsigned int tile_h,
                                            unsigned long long stride_elems) {
  unsigned long long ga = (unsigned long long)(uintptr_t)gaddr;
  u32x4 g0;
  g0[0] = 1u;                                   // count=1, user mode
  g0[1] = lds_off;                              // lds_addr (bytes)
  g0[2] = (unsigned int)ga;                     // global_addr[31:0]
  g0[3] = (unsigned int)((ga >> 32) & 0x01FFFFFFu) | 0x80000000u;  // [56:32] | type=2
  const unsigned int tw = 0x100000u, thh = 0x100000u;  // huge tensor dims: no clipping
  i32x8 g1;
  g1[0] = (int)(elem_code << 16);               // workgroup_mask=0, data_size
  g1[1] = (int)((tw & 0xFFFFu) << 16);          // tensor_dim0[15:0]
  g1[2] = (int)(((tw >> 16) & 0xFFFFu) | ((thh & 0xFFFFu) << 16));
  g1[3] = (int)(((thh >> 16) & 0xFFFFu) | (tile_w << 16));  // tile_dim0
  g1[4] = (int)(tile_h & 0xFFFFu);              // tile_dim1, tile_dim2=0
  g1[5] = (int)(stride_elems & 0xFFFFFFFFull);  // tensor_dim0_stride[31:0]
  g1[6] = (int)((stride_elems >> 32) & 0xFFFFull);
  g1[7] = 0;
  i32x4 z4 = {0, 0, 0, 0};
#if __clang_major__ >= 23
  i32x8 z8 = {0, 0, 0, 0, 0, 0, 0, 0};
  __builtin_amdgcn_tensor_load_to_lds(g0, g1, z4, z4, z8, 0);
#else
  __builtin_amdgcn_tensor_load_to_lds(g0, g1, z4, z4, 0);
#endif
}
#endif

// ---------------------------------------------------------------------------
__global__ void cast_f32_to_f16_kernel(const float* __restrict__ in,
                                       _Float16* __restrict__ out, int n) {
  int i = blockIdx.x * blockDim.x + threadIdx.x;
  if (i < n) out[i] = (_Float16)in[i];
}

// Pack f32 weights into f16 WMMA B-operand order:
// Bp[((h*KT + kt)*CT + ct)*32 + lane][e] = W[h][kt*32 + b_kk(lane,e)][ct*16 + (lane&15)]
__global__ void pack_wb_kernel(const float* __restrict__ W,
                               _Float16* __restrict__ Bp,
                               int K, int NC, int total) {
  int idx = blockIdx.x * blockDim.x + threadIdx.x;
  if (idx >= total) return;
  int lane = idx & 31;
  int rest = idx >> 5;
  int CT = NC >> 4, KT = K >> 5;
  int ct = rest % CT; rest /= CT;
  int kt = rest % KT;
  int h  = rest / KT;
  const float* __restrict__ Wh_ = W + (size_t)h * K * NC;
  v16h v;
#pragma unroll
  for (int e = 0; e < 16; ++e)
    v[e] = (_Float16)Wh_[(size_t)(kt * 32 + b_kk(lane, e)) * NC + ct * 16 + (lane & 15)];
  *reinterpret_cast<v16h*>(Bp + (size_t)idx * 16) = v;
}

// Wh[h] = A(f16)[NROWS x K] @ W(packed B operands)[h].
// Writes Wh f32 row-major (for edge scores) AND Wh f16 in B-operand packed
// order (consumed directly as the attention WMMA B operand).
// grid = (NROWS/16, H), block = 32.
template <int NC>
__global__ void gemm_wh_kernel(const _Float16* __restrict__ A,
                               const _Float16* __restrict__ Bp,
                               float* __restrict__ Cf,
                               _Float16* __restrict__ Cpk,
                               int NROWS, int K) {
  constexpr int CT = NC / 16;
  const int lane = threadIdx.x;
  const int h    = blockIdx.y;
  const int row0 = blockIdx.x * 16;
  const int KT   = K >> 5;
  const int m    = lane & 15;
  const int kb   = (lane < 16) ? 0 : 8;
  const _Float16* __restrict__ Arow = A + (size_t)(row0 + m) * K;
  const _Float16* __restrict__ Bph  = Bp + ((size_t)h * KT) * (CT * 32 * 16);

  v8f acc[CT] = {};
  for (int kt = 0; kt < KT; ++kt) {
    v8h lo = *reinterpret_cast<const v8h*>(Arow + kt * 32 + kb);
    v8h hi = *reinterpret_cast<const v8h*>(Arow + kt * 32 + kb + 16);
    v16h a = __builtin_shufflevector(lo, hi, 0, 1, 2, 3, 4, 5, 6, 7,
                                     8, 9, 10, 11, 12, 13, 14, 15);
#pragma unroll
    for (int ct = 0; ct < CT; ++ct) {
      v16h b = *reinterpret_cast<const v16h*>(
          Bph + ((size_t)kt * CT + ct) * (32 * 16) + lane * 16);
      acc[ct] = __builtin_amdgcn_wmma_f32_16x16x32_f16(false, a, false, b,
                                                       (short)0, acc[ct],
                                                       false, false);
    }
  }

  float*    Cfh = Cf + (size_t)h * NROWS * NC;
  // Packed-B epilogue addressing: output row j -> (jt = j>>5, krel = j&31);
  // B lane = (o&15) + 16*((krel>>3)&1), elem = (krel&7) + ((krel&16)>>1).
  const int jt    = row0 >> 5;
  const int khalf = row0 & 16;
  _Float16* Cph = Cpk + (((size_t)h * (NROWS >> 5) + jt) * CT) * (32 * 16);
#pragma unroll
  for (int ct = 0; ct < CT; ++ct) {
#pragma unroll
    for (int r = 0; r < 8; ++r) {
      int mr = ((lane < 16) ? 0 : 8) + r;
      float v = acc[ct][r];
      Cfh[(size_t)(row0 + mr) * NC + (ct * 16 + (lane & 15))] = v;
      int lb = (lane & 15) + ((mr & 8) << 1);   // B-operand lane
      int e  = (mr & 7) + (khalf >> 1);         // B-operand element (krel = khalf + mr)
      Cph[(size_t)ct * (32 * 16) + lb * 16 + e] = (_Float16)v;
    }
  }
}

// e_src/e_dst = Wh . a_{src,dst}; grid = (ceil(N/256), H), block = 256.
__global__ void edge_scores_kernel(const float* __restrict__ wh,
                                   const float* __restrict__ a,
                                   float* __restrict__ esrc,
                                   float* __restrict__ edst,
                                   int N, int OUTF) {
  int h = blockIdx.y;
  int n = blockIdx.x * blockDim.x + threadIdx.x;
  if (n >= N) return;
  const float* __restrict__ whr = wh + ((size_t)h * N + n) * OUTF;
  const float* __restrict__ ah  = a + (size_t)h * 2 * OUTF;
  float s0 = 0.f, s1 = 0.f;
  for (int o = 0; o < OUTF; ++o) {
    float w = whr[o];
    s0 += w * ah[o];
    s1 += w * ah[OUTF + o];
  }
  esrc[(size_t)h * N + n] = s0;
  edst[(size_t)h * N + n] = s1;
}

// Per-head max of e_dst. grid = H, block = 256.
__global__ void head_max_kernel(const float* __restrict__ edst,
                                float* __restrict__ dmax, int N) {
  int h = blockIdx.x, t = threadIdx.x;
  __shared__ float sm[256];
  float mx = -3.0e38f;
  for (int n = t; n < N; n += 256) mx = fmaxf(mx, edst[(size_t)h * N + n]);
  sm[t] = mx;
  __syncthreads();
  for (int s = 128; s > 0; s >>= 1) {
    if (t < s) sm[t] = fmaxf(sm[t], sm[t + s]);
    __syncthreads();
  }
  if (t == 0) dmax[h] = sm[0];
}

// Fused masked-softmax attention + aggregation.
// grid = (N/64, H), block = 128 (4 waves; wave w owns rows [i0, i0+16)).
// adj: double-buffered TDM staging (tile t+1 DMAs while tile t computes).
// Wh: packed B operands loaded directly from global (L2-resident).
template <int OUTF, bool CONCAT>
__global__ void gat_attn_kernel(const float* __restrict__ adj,
                                const float* __restrict__ esrc,
                                const float* __restrict__ edst,
                                const float* __restrict__ dmax,
                                const _Float16* __restrict__ whpk,
                                float* __restrict__ outh,      // !CONCAT: [H][N][OUTF]
                                _Float16* __restrict__ h1out,  // CONCAT : [N][H*OUTF]
                                int N) {
  constexpr int NT = OUTF / 16;
  const int tid  = threadIdx.x;
  const int w    = tid >> 5;
  const int lane = tid & 31;
  const int h    = blockIdx.y;
  const int i0   = blockIdx.x * 64 + w * 16;
  const int m    = lane & 15;
  const int kb   = (lane < 16) ? 0 : 8;
  const int row  = i0 + m;

  __shared__ __attribute__((aligned(16))) float sh_adj[2][64 * 32];
  __shared__ __attribute__((aligned(16))) float sh_edst[2][32];
  __shared__ float sh_z[4][16];

  const _Float16* __restrict__ wpk = whpk + ((size_t)h * (N >> 5) * NT) * (32 * 16);
  const float* __restrict__ edh    = edst + (size_t)h * N;
  const float es = esrc[(size_t)h * N + row];
  // Safe softmax shift: lrelu monotone => lrelu(es + max_j edst) >= row max.
  float mt = es + dmax[h];
  const float mrow = (mt > 0.f) ? mt : GAT_ALPHA * mt;

  auto stage = [&](int j0, int buf) {
#ifdef GAT_HAVE_TDM
    // Wave w DMAs its own 16x32 f32 adj tile.
    tdm_load_2d((unsigned int)(uintptr_t)&sh_adj[buf][w * 16 * 32],
                adj + (size_t)i0 * N + j0, /*4B*/ 2, 32, 16,
                (unsigned long long)N);
#else
    for (int idx = tid; idx < 64 * 32; idx += 128)
      sh_adj[buf][idx] =
          adj[(size_t)(blockIdx.x * 64 + (idx >> 5)) * N + j0 + (idx & 31)];
#endif
    if (tid < 32) sh_edst[buf][tid] = edh[j0 + tid];
  };

  float zacc = 0.f;
  v8f acc[NT] = {};

  stage(0, 0);  // prologue: tile 0 in flight
  const int niter = N / 32;
  for (int it = 0; it < niter; ++it) {
    const int c = it & 1;
    __syncthreads();  // iter it-1 LDS reads complete -> buffer c^1 reusable
    const bool more = (it + 1) < niter;
    if (more) stage((it + 1) * 32, c ^ 1);
#ifdef GAT_HAVE_TDM
    // 1 TDM op/tile, in-order per wave: <=1 outstanding == current tile done.
    if (more) __builtin_amdgcn_s_wait_tensorcnt((short)1);
    else      __builtin_amdgcn_s_wait_tensorcnt((short)0);
#endif
    __syncthreads();  // staged tile visible to all waves

    // adj + e_dst: two contiguous 8-float runs per lane -> b128 LDS loads.
    const float4* ar0 = reinterpret_cast<const float4*>(
        &sh_adj[c][(w * 16 + m) * 32 + kb]);
    const float4* ar1 = reinterpret_cast<const float4*>(
        &sh_adj[c][(w * 16 + m) * 32 + kb + 16]);
    const float4* er0 = reinterpret_cast<const float4*>(&sh_edst[c][kb]);
    const float4* er1 = reinterpret_cast<const float4*>(&sh_edst[c][kb + 16]);
    float4 a0 = ar0[0], a1 = ar0[1], a2 = ar1[0], a3 = ar1[1];
    float4 e0 = er0[0], e1 = er0[1], e2 = er1[0], e3 = er1[1];
    float adv[16] = {a0.x, a0.y, a0.z, a0.w, a1.x, a1.y, a1.z, a1.w,
                     a2.x, a2.y, a2.z, a2.w, a3.x, a3.y, a3.z, a3.w};
    float edv[16] = {e0.x, e0.y, e0.z, e0.w, e1.x, e1.y, e1.z, e1.w,
                     e2.x, e2.y, e2.z, e2.w, e3.x, e3.y, e3.z, e3.w};

    v16h aP;
#pragma unroll
    for (int e = 0; e < 16; ++e) {
      float ev = es + edv[e];
      ev = (ev > 0.f) ? ev : GAT_ALPHA * ev;
      float p = (adv[e] != 0.f) ? __expf(ev - mrow) : 0.f;
      zacc += p;
      aP[e] = (_Float16)p;
    }
    // B operands: contiguous 32B per lane from the packed Wh (L2-resident).
    const _Float16* __restrict__ bbase = wpk + ((size_t)it * NT) * (32 * 16);
#pragma unroll
    for (int t = 0; t < NT; ++t) {
      v16h b = *reinterpret_cast<const v16h*>(bbase + t * (32 * 16) + lane * 16);
      acc[t] = __builtin_amdgcn_wmma_f32_16x16x32_f16(false, aP, false, b,
                                                      (short)0, acc[t],
                                                      false, false);
    }
  }

  // Row sums: lane l and l^16 cover disjoint k-ranges of the same row.
  zacc += __shfl_xor(zacc, 16, 32);
  if (lane < 16) sh_z[w][lane] = zacc;
  __syncthreads();

#pragma unroll
  for (int t = 0; t < NT; ++t) {
#pragma unroll
    for (int r = 0; r < 8; ++r) {
      int mr  = ((lane < 16) ? 0 : 8) + r;
      int col = t * 16 + (lane & 15);
      float z = sh_z[w][mr];
      float v = (z > 0.f) ? (acc[t][r] / z) : 0.f;  // fully-masked row -> 0
      int orow = i0 + mr;
      if (CONCAT) {
        float e = (v > 0.f) ? v : (__expf(v) - 1.f);  // ELU
        h1out[(size_t)orow * (GAT_H * OUTF) + h * OUTF + col] = (_Float16)e;
      } else {
        outh[((size_t)h * N + orow) * OUTF + col] = v;
      }
    }
  }
}

// Head-mean + ELU + LayerNorm. grid = N, block = 128.
__global__ void mean_elu_ln_kernel(const float* __restrict__ outh,
                                   const float* __restrict__ gamma,
                                   const float* __restrict__ beta,
                                   float* __restrict__ tmp, int N) {
  int n = blockIdx.x, o = threadIdx.x;
  float s = 0.f;
#pragma unroll
  for (int h = 0; h < GAT_H; ++h)
    s += outh[((size_t)h * N + n) * GAT_EMB + o];
  s *= (1.0f / GAT_H);
  s = (s > 0.f) ? s : (__expf(s) - 1.f);  // ELU

  __shared__ float red[GAT_EMB];
  red[o] = s;
  __syncthreads();
  for (int st = GAT_EMB / 2; st > 0; st >>= 1) {
    if (o < st) red[o] += red[o + st];
    __syncthreads();
  }
  float mu = red[0] * (1.0f / GAT_EMB);
  __syncthreads();
  float d = s - mu;
  red[o] = d * d;
  __syncthreads();
  for (int st = GAT_EMB / 2; st > 0; st >>= 1) {
    if (o < st) red[o] += red[o + st];
    __syncthreads();
  }
  float var = red[0] * (1.0f / GAT_EMB);
  tmp[(size_t)n * GAT_EMB + o] = d * rsqrtf(var + 1e-5f) * gamma[o] + beta[o];
}

// Final mean over nodes. grid = 1, block = 128.
__global__ void col_mean_kernel(const float* __restrict__ tmp,
                                float* __restrict__ out, int N) {
  int o = threadIdx.x;
  float s = 0.f;
  for (int n = 0; n < N; ++n) s += tmp[(size_t)n * GAT_EMB + o];
  out[o] = s / (float)N;
}

// ---------------------------------------------------------------------------
extern "C" void kernel_launch(void* const* d_in, const int* in_sizes, int n_in,
                              void* d_out, int out_size, void* d_ws, size_t ws_size,
                              hipStream_t stream) {
  (void)in_sizes; (void)n_in; (void)out_size; (void)ws_size;
  const int N = GAT_N;

  const float* adj   = (const float*)d_in[0];
  const float* feats = (const float*)d_in[1];
  const float* W0    = (const float*)d_in[2];  // [H,128,64]
  const float* a0    = (const float*)d_in[3];  // [H,128]
  const float* W1    = (const float*)d_in[4];  // [H,256,128]
  const float* a1    = (const float*)d_in[5];  // [H,256]
  const float* gamma = (const float*)d_in[6];
  const float* beta  = (const float*)d_in[7];
  float* outp = (float*)d_out;

  // Workspace layout (~33 MB), 256B aligned slices.
  char* ws = (char*)d_ws;
  size_t off = 0;
  auto alloc = [&](size_t bytes) -> char* {
    char* p = ws + off;
    off = (off + bytes + 255) & ~(size_t)255;
    return p;
  };
  _Float16* feat16 = (_Float16*)alloc((size_t)N * GAT_FIN * 2);
  _Float16* w0_pk  = (_Float16*)alloc((size_t)GAT_H * GAT_FIN * 64 * 2);
  _Float16* w1_pk  = (_Float16*)alloc((size_t)GAT_H * 256 * GAT_EMB * 2);
  float*    wh0f   = (float*)   alloc((size_t)GAT_H * N * 64 * 4);
  _Float16* wh0pk  = (_Float16*)alloc((size_t)GAT_H * N * 64 * 2);
  float*    esrc0  = (float*)   alloc((size_t)GAT_H * N * 4);
  float*    edst0  = (float*)   alloc((size_t)GAT_H * N * 4);
  float*    dmax0  = (float*)   alloc(256);
  _Float16* h1_16  = (_Float16*)alloc((size_t)N * 256 * 2);
  float*    wh1f   = (float*)   alloc((size_t)GAT_H * N * 128 * 4);
  _Float16* wh1pk  = (_Float16*)alloc((size_t)GAT_H * N * 128 * 2);
  float*    esrc1  = (float*)   alloc((size_t)GAT_H * N * 4);
  float*    edst1  = (float*)   alloc((size_t)GAT_H * N * 4);
  float*    dmax1  = (float*)   alloc(256);
  float*    outh   = (float*)   alloc((size_t)GAT_H * N * 128 * 4);
  float*    tmp    = (float*)   alloc((size_t)N * GAT_EMB * 4);

  // --- input prep: feature cast + weight packing into B-operand order ---
  {
    int n = N * GAT_FIN;
    cast_f32_to_f16_kernel<<<(n + 255) / 256, 256, 0, stream>>>(feats, feat16, n);
    int t0 = GAT_H * (GAT_FIN / 32) * (64 / 16) * 32;   // 2048
    pack_wb_kernel<<<(t0 + 255) / 256, 256, 0, stream>>>(W0, w0_pk, GAT_FIN, 64, t0);
    int t1 = GAT_H * (256 / 32) * (GAT_EMB / 16) * 32;  // 8192
    pack_wb_kernel<<<(t1 + 255) / 256, 256, 0, stream>>>(W1, w1_pk, 256, GAT_EMB, t1);
  }

  // --- layer 0 ---
  gemm_wh_kernel<64><<<dim3(N / 16, GAT_H), 32, 0, stream>>>(
      feat16, w0_pk, wh0f, wh0pk, N, GAT_FIN);
  edge_scores_kernel<<<dim3((N + 255) / 256, GAT_H), 256, 0, stream>>>(
      wh0f, a0, esrc0, edst0, N, 64);
  head_max_kernel<<<GAT_H, 256, 0, stream>>>(edst0, dmax0, N);
  gat_attn_kernel<64, true><<<dim3(N / 64, GAT_H), 128, 0, stream>>>(
      adj, esrc0, edst0, dmax0, wh0pk, nullptr, h1_16, N);

  // --- layer 1 ---
  gemm_wh_kernel<128><<<dim3(N / 16, GAT_H), 32, 0, stream>>>(
      h1_16, w1_pk, wh1f, wh1pk, N, 256);
  edge_scores_kernel<<<dim3((N + 255) / 256, GAT_H), 256, 0, stream>>>(
      wh1f, a1, esrc1, edst1, N, 128);
  head_max_kernel<<<GAT_H, 256, 0, stream>>>(edst1, dmax1, N);
  gat_attn_kernel<128, false><<<dim3(N / 64, GAT_H), 128, 0, stream>>>(
      adj, esrc1, edst1, dmax1, wh1pk, outh, nullptr, N);

  // --- epilogue ---
  mean_elu_ln_kernel<<<N, GAT_EMB, 0, stream>>>(outh, gamma, beta, tmp, N);
  col_mean_kernel<<<1, GAT_EMB, 0, stream>>>(tmp, outp, N);
}
